// ArcConceptDecoder_55250459296392
// MI455X (gfx1250) — matrix operand order repaired and checked by reading, not compiled, not run on hardware
//
#include <hip/hip_runtime.h>
#include <math.h>

// ---------------- problem constants ----------------
#define SRC_  512
#define BSZ_  16
#define SNT_  512
#define EDIM  512
#define CDIM  512
#define VDIM  12000
#define EXT_  500
#define ROWS  (SRC_ * BSZ_)     // 8192
#define TOTE  (VDIM + EXT_)     // 12500
#define KDIM  512               // shared K for both GEMMs

// ---------------- WMMA types (CDNA5 / gfx1250, wave32) ----------------
typedef __attribute__((ext_vector_type(16))) __bf16 v16bf;
typedef __attribute__((ext_vector_type(8)))  float  v8f;

union FragBF { int4 q[2]; v16bf v; };   // 32 bytes: two b128 LDS loads -> one WMMA operand

static __device__ __forceinline__ unsigned short f32_to_bf16(float x) {
  unsigned int u = __float_as_uint(x);
  u += 0x7fffu + ((u >> 16) & 1u);      // round-to-nearest-even
  return (unsigned short)(u >> 16);
}

// ---------------- f32 -> bf16 conversion ----------------
__global__ __launch_bounds__(256)
void cvt_bf16_kernel(const float* __restrict__ in, unsigned short* __restrict__ out, int n) {
  int i = blockIdx.x * 256 + threadIdx.x;
  if (i < n) out[i] = f32_to_bf16(in[i]);
}

// ---------------- WMMA GEMM: out[M,N] = A[M,K] * Bm[N,K]^T + bias ----------------
// Block: 256 threads = 8 waves (2M x 4N), block tile 64M x 128N,
// wave tile 32x32 = four 16x16 WMMA accumulators.
// Double-buffered LDS + register prefetch: one barrier per K-step.
// MODE 0: h = tanh(.), write f32 (ld=512) + bf16 copy
// MODE 1: logits, write f32 into d_out rows of stride ldOut (=12500); N padded, stores guarded
template<int MODE>
__global__ __launch_bounds__(256)
void gemm_wmma_kernel(const unsigned short* __restrict__ A,   // M x K bf16 row-major
                      const unsigned short* __restrict__ Bm,  // N x K bf16 row-major (i.e. B^T)
                      const float* __restrict__ bias,         // Ntot
                      float* __restrict__ outF,
                      unsigned short* __restrict__ outH,
                      int Ntot, int ldOut)
{
  __shared__ __align__(16) unsigned short As[2][64 * 32];   // 2 x 4 KB
  __shared__ __align__(16) unsigned short Bs[2][128 * 32];  // 2 x 8 KB

  const int tid    = threadIdx.x;
  const int lane   = tid & 31;
  const int wave   = tid >> 5;
  const int wave_m = wave >> 2;        // 0..1 -> M offset *32
  const int wave_n = wave & 3;         // 0..3 -> N offset *32
  const int half   = lane >> 4;        // lane half selects K-range per ISA layout
  const int l15    = lane & 15;
  const int mbase  = blockIdx.y * 64;
  const int nbase  = blockIdx.x * 128;

  // staging assignment: thread stages one A b128 and two B b128 per K-step
  const int srow = tid >> 2;           // 0..63
  const int spart = (tid & 3) * 8;     // bf16 offset of 16B segment within 64B row
  const int brow0 = min(nbase + srow,      Ntot - 1);   // clamped (N padding)
  const int brow1 = min(nbase + srow + 64, Ntot - 1);

  v8f acc[2][2];
#pragma unroll
  for (int mi = 0; mi < 2; ++mi)
#pragma unroll
    for (int ni = 0; ni < 2; ++ni)
      acc[mi][ni] = (v8f){0.f, 0.f, 0.f, 0.f, 0.f, 0.f, 0.f, 0.f};

  int4 ra, rb0, rb1;
  auto load_stage = [&](int kb) {
    ra  = *(const int4*)&A [(size_t)(mbase + srow) * KDIM + kb + spart];
    rb0 = *(const int4*)&Bm[(size_t)brow0 * KDIM + kb + spart];
    rb1 = *(const int4*)&Bm[(size_t)brow1 * KDIM + kb + spart];
  };
  auto store_stage = [&](int p) {
    *(int4*)&As[p][srow * 32 + spart]        = ra;
    *(int4*)&Bs[p][srow * 32 + spart]        = rb0;
    *(int4*)&Bs[p][(srow + 64) * 32 + spart] = rb1;
  };

  load_stage(0);
  store_stage(0);

  int p = 0;
  for (int kb = 0; kb < KDIM; kb += 32) {
    __syncthreads();   // stores into buf p visible; everyone done reading buf p^1
    const bool has_next = (kb + 32 < KDIM);
    if (has_next) load_stage(kb + 32);               // overlap with WMMAs below
    if (kb + 64 < KDIM) {                            // pull 2 steps ahead toward WGP
      __builtin_prefetch(&A [(size_t)(mbase + srow) * KDIM + kb + 64 + spart], 0, 0);
      __builtin_prefetch(&Bm[(size_t)brow0 * KDIM + kb + 64 + spart], 0, 0);
      __builtin_prefetch(&Bm[(size_t)brow1 * KDIM + kb + 64 + spart], 0, 0);
    }

    // A fragments: lane<16 -> K{0..7,16..23}, lane>=16 -> K{8..15,24..31}
    FragBF a[2], b[2];
#pragma unroll
    for (int mi = 0; mi < 2; ++mi) {
      const unsigned short* ap = &As[p][(wave_m * 32 + mi * 16 + l15) * 32];
      a[mi].q[0] = *(const int4*)&ap[half * 8];
      a[mi].q[1] = *(const int4*)&ap[half * 8 + 16];
    }
    // B fragments: lane<16 -> K{0..7,16..23}, lane>=16 -> K{8..15,24..31} (column = l15)
#pragma unroll
    for (int ni = 0; ni < 2; ++ni) {
      const unsigned short* bp = &Bs[p][(wave_n * 32 + ni * 16 + l15) * 32];
      b[ni].q[0] = *(const int4*)&bp[half * 8];
      b[ni].q[1] = *(const int4*)&bp[half * 8 + 16];
    }

#pragma unroll
    for (int mi = 0; mi < 2; ++mi)
#pragma unroll
      for (int ni = 0; ni < 2; ++ni)
        acc[mi][ni] = __builtin_amdgcn_wmma_f32_16x16x32_bf16(
            /*neg_a=*/false, a[mi].v, /*neg_b=*/false, b[ni].v,
            /*c_mod=*/(short)0, acc[mi][ni], /*reuse_a=*/false, /*reuse_b=*/false);

    if (has_next) store_stage(p ^ 1);  // safe: barrier above proved p^1 fully consumed
    p ^= 1;
  }

  // C/D layout: VGPR v -> M = v + half*8 ; N = l15
#pragma unroll
  for (int mi = 0; mi < 2; ++mi)
#pragma unroll
    for (int ni = 0; ni < 2; ++ni) {
      const int col = nbase + wave_n * 32 + ni * 16 + l15;
      if (col < Ntot) {
        const float bv = bias[col];
#pragma unroll
        for (int v = 0; v < 8; ++v) {
          int mrow = mbase + wave_m * 32 + mi * 16 + v + half * 8;
          float x = acc[mi][ni][v] + bv;
          if (MODE == 0) {
            x = tanhf(x);
            outF[(size_t)mrow * ldOut + col] = x;
            outH[(size_t)mrow * KDIM  + col] = f32_to_bf16(x);
          } else {
            outF[(size_t)mrow * ldOut + col] = x;
          }
        }
      }
    }
}

// ---------------- gates = softmax(h @ W_div^T + b_div), one wave per row ----------------
__global__ __launch_bounds__(32)
void gates_kernel(const float* __restrict__ h, const float* __restrict__ Wdiv,
                  const float* __restrict__ bdiv, float* __restrict__ gates)
{
  int r = blockIdx.x, lane = threadIdx.x;
  float s0 = 0.f, s1 = 0.f, s2 = 0.f;
  for (int k = lane; k < CDIM; k += 32) {
    float hv = h[(size_t)r * CDIM + k];
    s0 += hv * Wdiv[0 * CDIM + k];
    s1 += hv * Wdiv[1 * CDIM + k];
    s2 += hv * Wdiv[2 * CDIM + k];
  }
  for (int off = 16; off; off >>= 1) {
    s0 += __shfl_down(s0, off, 32);
    s1 += __shfl_down(s1, off, 32);
    s2 += __shfl_down(s2, off, 32);
  }
  if (lane == 0) {
    s0 += bdiv[0]; s1 += bdiv[1]; s2 += bdiv[2];
    float m  = fmaxf(s0, fmaxf(s1, s2));
    float e0 = __expf(s0 - m), e1 = __expf(s1 - m), e2 = __expf(s2 - m);
    float inv = 1.f / (e0 + e1 + e2);
    gates[r * 3 + 0] = e0 * inv;   // gen
    gates[r * 3 + 1] = e1 * inv;   // map
    gates[r * 3 + 2] = e2 * inv;   // copy
  }
}

// ---------------- per-row softmax * gen_gate over 12000 logits (in d_out), zero ext ----------------
__global__ __launch_bounds__(512)
void softmax_scale_kernel(float* __restrict__ out, const float* __restrict__ gates)
{
  __shared__ float rowbuf[VDIM];     // 48 KB of the 320 KB WGP LDS
  __shared__ float red[16];
  __shared__ float sM, sS;
  const int r = blockIdx.x, tid = threadIdx.x;
  float* rowp = out + (size_t)r * TOTE;

  float mx = -1e30f;
  for (int i = tid; i < VDIM; i += 512) {
    float x = rowp[i];
    rowbuf[i] = x;
    mx = fmaxf(mx, x);
  }
  for (int off = 16; off; off >>= 1) mx = fmaxf(mx, __shfl_xor(mx, off, 32));
  if ((tid & 31) == 0) red[tid >> 5] = mx;
  __syncthreads();
  if (tid == 0) { float v = red[0]; for (int i = 1; i < 16; ++i) v = fmaxf(v, red[i]); sM = v; }
  __syncthreads();

  const float m = sM;
  float s = 0.f;
  for (int i = tid; i < VDIM; i += 512) {
    float e = __expf(rowbuf[i] - m);
    rowbuf[i] = e;
    s += e;
  }
  for (int off = 16; off; off >>= 1) s += __shfl_xor(s, off, 32);
  if ((tid & 31) == 0) red[tid >> 5] = s;
  __syncthreads();
  if (tid == 0) { float v = 0.f; for (int i = 0; i < 16; ++i) v += red[i]; sS = v; }
  __syncthreads();

  const float scale = gates[r * 3 + 0] / sS;   // gen_gate / sum
  for (int i = tid; i < VDIM; i += 512) rowp[i] = rowbuf[i] * scale;
  for (int i = VDIM + tid; i < TOTE; i += 512) rowp[i] = 0.f;   // extension slots
}

// ---------------- scatter-add copy probabilities (duplicate indices -> atomics) ----------------
__global__ __launch_bounds__(256)
void scatter_kernel(float* __restrict__ out, const float* __restrict__ gates,
                    const float* __restrict__ align_w, const int* __restrict__ copy_seq)
{
  const int r = blockIdx.x;
  const int b = r & (BSZ_ - 1);
  const float gmap  = gates[r * 3 + 1];
  const float gcopy = gates[r * 3 + 2];
  float* rowp = out + (size_t)r * TOTE;
  for (int p = threadIdx.x; p < SNT_ * 2; p += 256) {
    int t = p >> 1, j = p & 1;
    int idx = copy_seq[(t * BSZ_ + b) * 2 + j];
    float val = (j ? gmap : gcopy) * align_w[(size_t)r * SNT_ + t];  // copy_gate2 = [copy, map]
    atomicAdd(rowp + idx, val);
  }
}

// ---------------- final log over ll, plus arc_ll ----------------
__global__ __launch_bounds__(256)
void log_kernel(float* __restrict__ out, const float* __restrict__ arc,
                size_t n_ll, size_t n_arc)
{
  size_t i = (size_t)blockIdx.x * 256 + threadIdx.x;
  size_t stride = (size_t)gridDim.x * 256;
  for (size_t k = i; k < n_ll; k += stride)
    out[k] = __logf(out[k] + 1e-12f);
  for (size_t k = i; k < n_arc; k += stride)
    out[n_ll + k] = __logf(arc[k] + 1e-12f);
}

// ---------------- launch ----------------
extern "C" void kernel_launch(void* const* d_in, const int* in_sizes, int n_in,
                              void* d_out, int out_size, void* d_ws, size_t ws_size,
                              hipStream_t stream) {
  const float* alignment_weight = (const float*)d_in[0];   // (512,16,512)
  const float* arc_weight       = (const float*)d_in[1];   // (512,16,512)
  const float* concept_outs     = (const float*)d_in[3];   // (512,16,512)
  const int*   copy_seq         = (const int*)d_in[4];     // (512,16,2)
  const float* W_transfer       = (const float*)d_in[9];   // (512,512)
  const float* b_transfer       = (const float*)d_in[10];  // (512,)
  const float* W_gen            = (const float*)d_in[11];  // (12000,512)
  const float* b_gen            = (const float*)d_in[12];  // (12000,)
  const float* W_div            = (const float*)d_in[13];  // (3,512)
  const float* b_div            = (const float*)d_in[14];  // (3,)
  float* out = (float*)d_out;

  // workspace layout (all 16B-aligned offsets)
  char* ws = (char*)d_ws;
  unsigned short* co_bf16 = (unsigned short*)(ws);                         // 8192*512*2  = 8,388,608
  unsigned short* wt_bf16 = (unsigned short*)(ws + 8388608);               // 512*512*2   =   524,288
  unsigned short* wg_bf16 = (unsigned short*)(ws + 8912896);               // 12000*512*2 = 12,288,000
  float*          h_f32   = (float*)         (ws + 21200896);              // 8192*512*4  = 16,777,216
  unsigned short* h_bf16  = (unsigned short*)(ws + 37978112);              // 8192*512*2  = 8,388,608
  float*          gates   = (float*)         (ws + 46366720);              // 8192*3*4

  // 1) f32 -> bf16 conversions
  cvt_bf16_kernel<<<(ROWS * EDIM + 255) / 256, 256, 0, stream>>>(concept_outs, co_bf16, ROWS * EDIM);
  cvt_bf16_kernel<<<(CDIM * EDIM + 255) / 256, 256, 0, stream>>>(W_transfer, wt_bf16, CDIM * EDIM);
  cvt_bf16_kernel<<<(VDIM * CDIM + 255) / 256, 256, 0, stream>>>(W_gen, wg_bf16, VDIM * CDIM);

  // 2) h = tanh(co @ Wt^T + b)   [8192 x 512], WMMA bf16
  {
    dim3 grid(CDIM / 128, ROWS / 64);          // (4, 128)
    gemm_wmma_kernel<0><<<grid, 256, 0, stream>>>(co_bf16, wt_bf16, b_transfer,
                                                  h_f32, h_bf16, CDIM, CDIM);
  }

  // 3) gates = softmax(h @ W_div^T + b_div)
  gates_kernel<<<ROWS, 32, 0, stream>>>(h_f32, W_div, b_div, gates);

  // 4) logits = h @ W_gen^T + b_gen, written into d_out rows (stride 12500), WMMA bf16
  {
    dim3 grid((VDIM + 127) / 128, ROWS / 64);  // (94, 128), N padded + guarded
    gemm_wmma_kernel<1><<<grid, 256, 0, stream>>>(h_bf16, wg_bf16, b_gen,
                                                  out, nullptr, VDIM, TOTE);
  }

  // 5) probs = gen_gate * softmax(logits); zero extension slots
  softmax_scale_kernel<<<ROWS, 512, 0, stream>>>(out, gates);

  // 6) scatter-add copy probabilities
  scatter_kernel<<<ROWS, 256, 0, stream>>>(out, gates, alignment_weight, copy_seq);

  // 7) ll = log(probs + 1e-12); arc_ll = log(arc_weight + 1e-12)
  {
    size_t n_ll  = (size_t)ROWS * TOTE;        // 102,400,000
    size_t n_arc = (size_t)ROWS * SNT_;        //   4,194,304
    log_kernel<<<8192, 256, 0, stream>>>(out, arc_weight, n_ll, n_arc);
  }
}
